// TemporalGraphTransformer_21861383537421
// MI455X (gfx1250) — matrix operand order repaired and checked by reading, not compiled, not run on hardware
//
#include <hip/hip_runtime.h>

// ============================================================================
// TemporalGraphTransformer forward for MI455X (gfx1250, wave32, WMMA).
// All dense GEMMs use v_wmma_f32_16x16x32_bf16 (bf16 inputs, f32 accumulate).
// B-matrices (weights) are pre-swizzled into WMMA fragment layout once per
// launch; A (activations) is loaded as f32 via b128 and packed to bf16 in
// registers. Graph scatter-softmax uses L2-resident f32/u32 atomics.
// Contract: all GEMMs have K % 32 == 0 and Nc % 32 == 0 (true for this model).
// ============================================================================

#define NN    10000      // nodes
#define TT    5          // timesteps
#define EE    100000     // edges
#define HIDD  128        // hidden
#define NHEAD 4          // conv heads
#define HC    512        // NHEAD*HIDD
#define NTOK  (NN*TT)    // transformer tokens

typedef __attribute__((ext_vector_type(16))) __bf16 v16bf;
typedef __attribute__((ext_vector_type(8)))  float  v8f;

enum { GF_BIAS = 1, GF_RELU = 2, GF_RES = 4 };

// ---------------------------------------------------------------------------
// order-preserving float <-> uint encoding for atomicMax-based segment max
// ---------------------------------------------------------------------------
__device__ __forceinline__ unsigned fenc(float f) {
  unsigned u = __float_as_uint(f);
  return (u & 0x80000000u) ? ~u : (u | 0x80000000u);
}
__device__ __forceinline__ float fdec(unsigned u) {
  return __uint_as_float((u & 0x80000000u) ? (u & 0x7FFFFFFFu) : ~u);
}

__global__ void fill_f32_kernel(float* __restrict__ p, float v, size_t n) {
  size_t i = (size_t)blockIdx.x * blockDim.x + threadIdx.x;
  if (i < n) p[i] = v;
}
__global__ void fill_u32_kernel(unsigned* __restrict__ p, unsigned v, size_t n) {
  size_t i = (size_t)blockIdx.x * blockDim.x + threadIdx.x;
  if (i < n) p[i] = v;
}

// ---------------------------------------------------------------------------
// Pre-swizzle a f32 weight matrix W[K x Nc] into bf16 WMMA B-fragments.
// Element (tileN, kblk, lane, j) = W[kblk*32 + (lane>>4)*16 + j, tileN*16 + (lane&15)]
// so in the GEMM each lane loads one contiguous 32B v16bf per K-step.
// ---------------------------------------------------------------------------
__global__ void pack_b_kernel(const float* __restrict__ W,
                              __bf16* __restrict__ out, int K, int Nc) {
  int idx = blockIdx.x * blockDim.x + threadIdx.x;
  if (idx >= K * Nc) return;
  int KB   = K >> 5;
  int j    = idx & 15;
  int lane = (idx >> 4) & 31;
  int t2   = idx >> 9;
  int kb   = t2 % KB;
  int tn   = t2 / KB;
  int k    = kb * 32 + (lane >> 4) * 16 + j;
  int col  = tn * 16 + (lane & 15);
  out[idx] = (__bf16)W[(size_t)k * Nc + col];
}

// ---------------------------------------------------------------------------
// WMMA bf16 GEMM:  C[M x Nc] = A[M x K] @ W[K x Nc] (+bias)(+res)(relu)
// One wave -> 16x32 output (two 16x16 N-tiles sharing the A fragment).
// No K/N guards (K%32==0, Nc%32==0); ragged M handled by row clamping on
// loads and a wave-uniform guarded store path.
// ---------------------------------------------------------------------------
__global__ __launch_bounds__(128) void
gemm_bf16_kernel(const float* __restrict__ A, const __bf16* __restrict__ Bp,
                 const float* __restrict__ bias, const float* __restrict__ res,
                 float* __restrict__ C, int M, int K, int Nc, int flags) {
  const int wave = threadIdx.x >> 5;
  const int lane = threadIdx.x & 31;
  const int hf   = lane >> 4;              // wave half
  const int r    = lane & 15;

  const int row0 = (blockIdx.y * 4 + wave) * 16;
  if (row0 >= M) return;                   // wave-uniform: EXEC all-ones at WMMA
  const int tn0  = blockIdx.x * 2;         // two 16-col tiles
  const int n0   = tn0 * 16 + r;
  const int n1   = n0 + 16;

  const int m = min(row0 + r, M - 1);      // clamp: loads always valid
  const float4* __restrict__ Arow4 = (const float4*)(A + (size_t)m * K);
  const int KB = K >> 5;
  const v16bf* __restrict__ B0 = (const v16bf*)Bp + (size_t)tn0 * KB * 32 + lane;
  const v16bf* __restrict__ B1 = B0 + (size_t)KB * 32;

  v8f acc0 = {}, acc1 = {};
  int ai = hf * 4;                         // float4 index of K-16 block for this half
  for (int kb = 0; kb < KB; ++kb) {
    float4 p0 = Arow4[ai];                 // K = base .. +3
    float4 p1 = Arow4[ai + 1];             // K = +4 .. +7
    float4 p2 = Arow4[ai + 2];             // K = +8 .. +11
    float4 p3 = Arow4[ai + 3];             // K = +12 .. +15
    v16bf a;
    a[0]  = (__bf16)p0.x; a[1]  = (__bf16)p0.y; a[2]  = (__bf16)p0.z; a[3]  = (__bf16)p0.w;
    a[4]  = (__bf16)p1.x; a[5]  = (__bf16)p1.y; a[6]  = (__bf16)p1.z; a[7]  = (__bf16)p1.w;
    a[8]  = (__bf16)p2.x; a[9]  = (__bf16)p2.y; a[10] = (__bf16)p2.z; a[11] = (__bf16)p2.w;
    a[12] = (__bf16)p3.x; a[13] = (__bf16)p3.y; a[14] = (__bf16)p3.z; a[15] = (__bf16)p3.w;
    v16bf b0 = B0[(size_t)kb * 32];
    v16bf b1 = B1[(size_t)kb * 32];
    acc0 = __builtin_amdgcn_wmma_f32_16x16x32_bf16(false, a, false, b0, (short)0, acc0, false, false);
    acc1 = __builtin_amdgcn_wmma_f32_16x16x32_bf16(false, a, false, b1, (short)0, acc1, false, false);
    ai += 8;                               // advance 32 floats
  }

  const float bn0 = (flags & GF_BIAS) ? bias[n0] : 0.0f;
  const float bn1 = (flags & GF_BIAS) ? bias[n1] : 0.0f;
  if (row0 + 16 <= M) {                    // wave-uniform fast path
#pragma unroll
    for (int g = 0; g < 8; ++g) {          // D VGPR g -> row g + 8*hf
      size_t rowb = (size_t)(row0 + g + 8 * hf) * Nc;
      float v0 = acc0[g] + bn0, v1 = acc1[g] + bn1;
      if (flags & GF_RES) { v0 += res[rowb + n0]; v1 += res[rowb + n1]; }
      if (flags & GF_RELU) { v0 = fmaxf(v0, 0.0f); v1 = fmaxf(v1, 0.0f); }
      C[rowb + n0] = v0;
      C[rowb + n1] = v1;
    }
  } else {
#pragma unroll
    for (int g = 0; g < 8; ++g) {
      int rr = row0 + g + 8 * hf;
      if (rr < M) {
        size_t rowb = (size_t)rr * Nc;
        float v0 = acc0[g] + bn0, v1 = acc1[g] + bn1;
        if (flags & GF_RES) { v0 += res[rowb + n0]; v1 += res[rowb + n1]; }
        if (flags & GF_RELU) { v0 = fmaxf(v0, 0.0f); v1 = fmaxf(v1, 0.0f); }
        C[rowb + n0] = v0;
        C[rowb + n1] = v1;
      }
    }
  }
}

// ---------------------------------------------------------------------------
// concat([X_static | X_temp[t] | M_temp[t]]) -> [N,128]
// ---------------------------------------------------------------------------
__global__ void pack_xt_kernel(const float* __restrict__ Xs,
                               const float* __restrict__ Xt,
                               const float* __restrict__ Mt,
                               float* __restrict__ out) {
  size_t i = (size_t)blockIdx.x * blockDim.x + threadIdx.x;
  if (i >= (size_t)NN * HIDD) return;
  int nrow = (int)(i >> 7), c = (int)(i & 127);
  float v;
  if (c < 64)       v = Xs[(size_t)nrow * 64 + c];
  else if (c < 96)  v = Xt[(size_t)nrow * 32 + (c - 64)];
  else              v = Mt[(size_t)nrow * 32 + (c - 96)];
  out[i] = v;
}

// ---------------------------------------------------------------------------
// TransformerConv edge kernels. Edge projection e = eattr@We is recomputed
// in-register (K=8 dot) instead of materializing E*512 floats (205 MB).
// One block (4 heads x 32 lanes) per edge.
// ---------------------------------------------------------------------------
__global__ void edge_alpha_kernel(const float* __restrict__ q,
                                  const float* __restrict__ k,
                                  const float* __restrict__ eattr,
                                  const float* __restrict__ We,
                                  const int* __restrict__ src,
                                  const int* __restrict__ dst,
                                  float* __restrict__ alpha,
                                  unsigned* __restrict__ amax) {
  int e = blockIdx.x;
  int h = threadIdx.x >> 5, lane = threadIdx.x & 31;
  int sN = src[e], dN = dst[e];
  float eav[8];
#pragma unroll
  for (int j = 0; j < 8; ++j) eav[j] = eattr[(size_t)e * 8 + j];
  float sum = 0.0f;
#pragma unroll
  for (int jj = 0; jj < 4; ++jj) {
    int col = h * HIDD + jj * 32 + lane;
    float ev = 0.0f;
#pragma unroll
    for (int j = 0; j < 8; ++j) ev += eav[j] * We[j * HC + col];
    sum += q[(size_t)dN * HC + col] * (k[(size_t)sN * HC + col] + ev);
  }
#pragma unroll
  for (int o = 16; o; o >>= 1) sum += __shfl_xor(sum, o, 32);
  if (lane == 0) {
    float a = sum * 0.08838834764831845f;   // 1/sqrt(128)
    alpha[(size_t)e * NHEAD + h] = a;
    atomicMax(amax + (size_t)dN * NHEAD + h, fenc(a));
  }
}

__global__ void edge_exp_kernel(float* __restrict__ alpha,
                                const unsigned* __restrict__ amax,
                                const int* __restrict__ dst,
                                float* __restrict__ den) {
  int i = blockIdx.x * blockDim.x + threadIdx.x;
  if (i >= EE * NHEAD) return;
  int e = i >> 2, h = i & 3;
  int dN = dst[e];
  float ex = __expf(alpha[i] - fdec(amax[(size_t)dN * NHEAD + h]));
  alpha[i] = ex;                           // reuse buffer for exp values
  atomicAdd(den + (size_t)dN * NHEAD + h, ex);
}

__global__ void edge_msg_kernel(const float* __restrict__ v,
                                const float* __restrict__ eattr,
                                const float* __restrict__ We,
                                const float* __restrict__ ex,
                                const float* __restrict__ den,
                                const int* __restrict__ src,
                                const int* __restrict__ dst,
                                float* __restrict__ msg) {
  int e = blockIdx.x;
  int h = threadIdx.x >> 5, lane = threadIdx.x & 31;
  int sN = src[e], dN = dst[e];
  float a = ex[(size_t)e * NHEAD + h] / (den[(size_t)dN * NHEAD + h] + 1e-16f);
  float eav[8];
#pragma unroll
  for (int j = 0; j < 8; ++j) eav[j] = eattr[(size_t)e * 8 + j];
#pragma unroll
  for (int jj = 0; jj < 4; ++jj) {
    int col = h * HIDD + jj * 32 + lane;
    float ev = 0.0f;
#pragma unroll
    for (int j = 0; j < 8; ++j) ev += eav[j] * We[j * HC + col];
    atomicAdd(msg + (size_t)dN * HC + col,
              a * (v[(size_t)sN * HC + col] + ev));
  }
}

// ---------------------------------------------------------------------------
// sinusoidal time encoding [T,32]
// ---------------------------------------------------------------------------
__global__ void time_enc_kernel(const int* __restrict__ years,
                                float* __restrict__ te) {
  int i = threadIdx.x;
  if (i >= TT * 32) return;
  int t = i >> 5, d = i & 31;
  float rel = (float)(years[t] - 2022);
  float div = __powf(10000.0f, 2.0f * floorf((float)d * 0.5f) / 32.0f);
  float ang = rel / div;
  te[i] = (d & 1) ? __cosf(ang) : __sinf(ang);
}

// Hs[T,N,128] + tproj[t] broadcast -> x[N,T,128]
__global__ void add_time_transpose_kernel(const float* __restrict__ Hs,
                                          const float* __restrict__ tp,
                                          float* __restrict__ x) {
  size_t i = (size_t)blockIdx.x * blockDim.x + threadIdx.x;
  if (i >= (size_t)TT * NN * HIDD) return;
  int t = (int)(i / ((size_t)NN * HIDD));
  size_t rem = i - (size_t)t * NN * HIDD;
  int nrow = (int)(rem >> 7), c = (int)(rem & 127);
  x[((size_t)nrow * TT + t) * HIDD + c] = Hs[i] + tp[t * HIDD + c];
}

// ---------------------------------------------------------------------------
// self-attention over T=5 tokens, 4 heads x 32 dims; one block per node
// ---------------------------------------------------------------------------
__global__ void mha_attn_kernel(const float* __restrict__ qkv,
                                float* __restrict__ o) {
  __shared__ float sq[TT * HIDD], sk[TT * HIDD], sv[TT * HIDD];
  __shared__ float satt[4 * TT * TT];
  int n = blockIdx.x, tid = threadIdx.x;
  const float* base = qkv + (size_t)n * TT * 384;
  for (int i = tid; i < TT * HIDD; i += 128) {
    int t = i >> 7, c = i & 127;
    sq[i] = base[t * 384 + c];
    sk[i] = base[t * 384 + 128 + c];
    sv[i] = base[t * 384 + 256 + c];
  }
  __syncthreads();
  if (tid < 100) {                         // (h,t,s) score tasks
    int h = tid / 25, rem = tid % 25, t = rem / 5, s = rem % 5;
    float sum = 0.0f;
#pragma unroll
    for (int c = 0; c < 32; ++c)
      sum += sq[t * HIDD + h * 32 + c] * sk[s * HIDD + h * 32 + c];
    satt[tid] = sum * 0.17677669529663687f;  // 1/sqrt(32)
  }
  __syncthreads();
  if (tid < 20) {                          // softmax per (h,t)
    int b = tid * 5;
    float m = satt[b];
#pragma unroll
    for (int s = 1; s < 5; ++s) m = fmaxf(m, satt[b + s]);
    float sm = 0.0f;
#pragma unroll
    for (int s = 0; s < 5; ++s) { float e2 = __expf(satt[b + s] - m); satt[b + s] = e2; sm += e2; }
    float inv = 1.0f / sm;
#pragma unroll
    for (int s = 0; s < 5; ++s) satt[b + s] *= inv;
  }
  __syncthreads();
  for (int i = tid; i < TT * HIDD; i += 128) {
    int t = i >> 7, c = i & 127, h = c >> 5;
    float sum = 0.0f;
#pragma unroll
    for (int s = 0; s < 5; ++s)
      sum += satt[h * 25 + t * 5 + s] * sv[s * HIDD + c];
    o[(size_t)n * (TT * HIDD) + i] = sum;
  }
}

// LayerNorm over last dim (128); one wave per row, 8 rows per block
__global__ void layer_norm_kernel(const float* __restrict__ in,
                                  const float* __restrict__ g,
                                  const float* __restrict__ b,
                                  float* __restrict__ out, int rows) {
  int wv = threadIdx.x >> 5, lane = threadIdx.x & 31;
  int row = blockIdx.x * 8 + wv;
  if (row >= rows) return;
  const float* p = in + (size_t)row * HIDD;
  float x0 = p[lane], x1 = p[lane + 32], x2 = p[lane + 64], x3 = p[lane + 96];
  float s = x0 + x1 + x2 + x3;
#pragma unroll
  for (int o = 16; o; o >>= 1) s += __shfl_xor(s, o, 32);
  float mean = s * (1.0f / 128.0f);
  float d0 = x0 - mean, d1 = x1 - mean, d2 = x2 - mean, d3 = x3 - mean;
  float vs = d0 * d0 + d1 * d1 + d2 * d2 + d3 * d3;
#pragma unroll
  for (int o = 16; o; o >>= 1) vs += __shfl_xor(vs, o, 32);
  float inv = rsqrtf(vs * (1.0f / 128.0f) + 1e-5f);
  float* q = out + (size_t)row * HIDD;
  q[lane]      = d0 * inv * g[lane]      + b[lane];
  q[lane + 32] = d1 * inv * g[lane + 32] + b[lane + 32];
  q[lane + 64] = d2 * inv * g[lane + 64] + b[lane + 64];
  q[lane + 96] = d3 * inv * g[lane + 96] + b[lane + 96];
}

// decay weights; writes both scratch copy and the tuple tail of d_out
__global__ void compute_w_kernel(const int* __restrict__ years,
                                 float* __restrict__ wbuf,
                                 float* __restrict__ wout) {
  if (threadIdx.x == 0 && blockIdx.x == 0) {
    float ww[TT]; float s = 0.0f;
#pragma unroll
    for (int t = 0; t < TT; ++t) {
      float gap = fmaxf(0.0f, (float)(2022 - years[t]));
      ww[t] = __expf(-0.35f * gap);
      s += ww[t];
    }
    float inv = 1.0f / (s + 1e-8f);
#pragma unroll
    for (int t = 0; t < TT; ++t) { wbuf[t] = ww[t] * inv; wout[t] = ww[t] * inv; }
  }
}

__global__ void weighted_sum_kernel(const float* __restrict__ x,
                                    const float* __restrict__ w,
                                    float* __restrict__ z) {
  size_t i = (size_t)blockIdx.x * blockDim.x + threadIdx.x;
  if (i >= (size_t)NN * HIDD) return;
  int nrow = (int)(i >> 7), c = (int)(i & 127);
  float s = 0.0f;
#pragma unroll
  for (int t = 0; t < TT; ++t)
    s += w[t] * x[((size_t)nrow * TT + t) * HIDD + c];
  z[i] = s;
}

// ============================================================================
// host orchestration
// ============================================================================
extern "C" void kernel_launch(void* const* d_in, const int* in_sizes, int n_in,
                              void* d_out, int out_size, void* d_ws, size_t ws_size,
                              hipStream_t stream) {
  (void)in_sizes; (void)n_in; (void)out_size; (void)ws_size;

  // ---- top-level inputs: setup_inputs() insertion order ----
  const float* Xs   = (const float*)d_in[0];   // [N,64]
  const float* Xt   = (const float*)d_in[1];   // [T,N,32]
  const float* Mt   = (const float*)d_in[2];   // [T,N,32]
  const int*   yrs  = (const int*)d_in[3];     // [T]
  const int*   ei_d = (const int*)d_in[4];     // [2,E]
  const float* ea_d = (const float*)d_in[5];   // [E,8]
  const int*   ei_u = (const int*)d_in[6];
  const float* ea_u = (const float*)d_in[7];

  // ---- params pytree flattened with sorted dict keys ----
  // conv: +0 We, +1 Wk, +2 Wq, +3 Ws, +4 Wv, +5 bk, +6 bq, +7 bs, +8 bv
  const int CD = 8, CU = 17;
  const float* inW  = (const float*)d_in[26];  // input_proj W [128,128]
  const float* inB  = (const float*)d_in[27];
  // layers at 28+12L: +0 W1, +1 W2, +2 Wo, +3 Wqkv, +4 b1, +5 b2,
  //                   +6 beta1, +7 beta2, +8 bo, +9 bqkv, +10 g1, +11 g2
  const float* linW = (const float*)d_in[52];  // [1024,128]
  const float* linB = (const float*)d_in[53];
  const float* roW  = (const float*)d_in[54];  // [128,128]
  const float* roB  = (const float*)d_in[55];
  const float* tW   = (const float*)d_in[56];  // [32,128]
  const float* tB   = (const float*)d_in[57];

  // ---- workspace bump allocator ----
  char* wsb = (char*)d_ws;
  size_t off = 0;
  auto alloc_bytes = [&](size_t bytes) -> void* {
    off = (off + 255) & ~(size_t)255;
    void* p = (void*)(wsb + off);
    off += bytes;
    return p;
  };
  auto allocf  = [&](size_t n) -> float*  { return (float*)alloc_bytes(n * 4); };
  auto allocbf = [&](size_t n) -> __bf16* { return (__bf16*)alloc_bytes(n * 2); };

  // pre-swizzle weight into bf16 WMMA B-fragment layout
  auto pack_b = [&](const float* w, int K, int Nc) -> const __bf16* {
    __bf16* p = allocbf((size_t)K * Nc);
    int n = K * Nc;
    pack_b_kernel<<<dim3((unsigned)((n + 255) / 256)), dim3(256), 0, stream>>>(w, p, K, Nc);
    return p;
  };

  const __bf16* Wq_d  = pack_b((const float*)d_in[CD + 2], 128, 512);
  const __bf16* Wk_d  = pack_b((const float*)d_in[CD + 1], 128, 512);
  const __bf16* Wv_d  = pack_b((const float*)d_in[CD + 4], 128, 512);
  const __bf16* Ws_d  = pack_b((const float*)d_in[CD + 3], 128, 512);
  const __bf16* Wq_u  = pack_b((const float*)d_in[CU + 2], 128, 512);
  const __bf16* Wk_u  = pack_b((const float*)d_in[CU + 1], 128, 512);
  const __bf16* Wv_u  = pack_b((const float*)d_in[CU + 4], 128, 512);
  const __bf16* Ws_u  = pack_b((const float*)d_in[CU + 3], 128, 512);
  const __bf16* Wi_pk = pack_b(inW, 128, 128);
  const __bf16* Wl_lo = pack_b(linW, 512, 128);                       // rows 0..511
  const __bf16* Wl_hi = pack_b(linW + (size_t)HC * HIDD, 512, 128);   // rows 512..1023
  const __bf16* Wt_pk = pack_b(tW, 32, 128);
  const __bf16* Wr_pk = pack_b(roW, 128, 128);
  const __bf16 *W1_pk[2], *W2_pk[2], *Wo_pk[2], *Wqkv_pk[2];
  for (int L = 0; L < 2; ++L) {
    int bidx = 28 + 12 * L;
    W1_pk[L]   = pack_b((const float*)d_in[bidx + 0], 128, 256);
    W2_pk[L]   = pack_b((const float*)d_in[bidx + 1], 256, 128);
    Wo_pk[L]   = pack_b((const float*)d_in[bidx + 2], 128, 128);
    Wqkv_pk[L] = pack_b((const float*)d_in[bidx + 3], 128, 384);
  }

  // persistent activations
  float*    xcat = allocf((size_t)NN * HIDD);
  float*    xt   = allocf((size_t)NN * HIDD);
  float*    exb  = allocf((size_t)EE * NHEAD);
  unsigned* amax = (unsigned*)alloc_bytes((size_t)NN * NHEAD * 4);
  float*    den  = allocf((size_t)NN * NHEAD);
  float*    Hs   = allocf((size_t)TT * NN * HIDD);
  float*    xb   = allocf((size_t)NTOK * HIDD);
  float*    te   = allocf(TT * 32);
  float*    tpj  = allocf(TT * HIDD);
  float*    wbuf = allocf(TT);
  float*    zb   = allocf((size_t)NN * HIDD);

  // phase-overlaid region: conv buffers vs transformer buffers
  size_t mark = off;
  float* qb  = allocf((size_t)NN * HC);
  float* kb  = allocf((size_t)NN * HC);
  float* vb  = allocf((size_t)NN * HC);
  float* msg = allocf((size_t)NN * HC);
  float* hd  = allocf((size_t)NN * HC);
  float* hu  = allocf((size_t)NN * HC);
  size_t convEnd = off;
  off = mark;                                  // transformer phase reuses it
  float* qkvb  = allocf((size_t)NTOK * 384);
  float* attno = allocf((size_t)NTOK * HIDD);
  float* ffh   = allocf((size_t)NTOK * 256);
  float* tmpb  = allocf((size_t)NTOK * HIDD);
  if (off < convEnd) off = convEnd;

  auto gemm = [&](const float* A, const __bf16* Bp, const float* bias,
                  const float* res, float* C, int M, int K, int Nc, int flags) {
    dim3 grid((unsigned)(Nc / 32), (unsigned)(((M + 15) / 16 + 3) / 4));
    gemm_bf16_kernel<<<grid, dim3(128), 0, stream>>>(A, Bp, bias, res, C, M, K, Nc, flags);
  };
  auto fillf = [&](float* p, float v, size_t n) {
    fill_f32_kernel<<<dim3((unsigned)((n + 255) / 256)), dim3(256), 0, stream>>>(p, v, n);
  };
  auto fillu = [&](unsigned* p, unsigned v, size_t n) {
    fill_u32_kernel<<<dim3((unsigned)((n + 255) / 256)), dim3(256), 0, stream>>>(p, v, n);
  };

  auto run_conv = [&](const __bf16* Wq, const float* bq, const __bf16* Wk,
                      const float* bk, const __bf16* Wv, const float* bv,
                      const __bf16* Ws, const float* bs, const float* We,
                      const int* src, const int* dst, const float* eattr,
                      float* hout) {
    gemm(xt, Wq, bq, nullptr, qb, NN, HIDD, HC, GF_BIAS);
    gemm(xt, Wk, bk, nullptr, kb, NN, HIDD, HC, GF_BIAS);
    gemm(xt, Wv, bv, nullptr, vb, NN, HIDD, HC, GF_BIAS);
    fillu(amax, 0u, (size_t)NN * NHEAD);        // 0 < fenc(any finite float)
    fillf(den, 0.0f, (size_t)NN * NHEAD);
    fillf(msg, 0.0f, (size_t)NN * HC);
    edge_alpha_kernel<<<dim3(EE), dim3(128), 0, stream>>>(qb, kb, eattr, We, src, dst, exb, amax);
    edge_exp_kernel<<<dim3((EE * NHEAD + 255) / 256), dim3(256), 0, stream>>>(exb, amax, dst, den);
    edge_msg_kernel<<<dim3(EE), dim3(128), 0, stream>>>(vb, eattr, We, exb, den, src, dst, msg);
    gemm(xt, Ws, bs, msg, hout, NN, HIDD, HC, GF_BIAS | GF_RES);  // skip + messages
  };

  // -------- temporal scan --------
  for (int t = 0; t < TT; ++t) {
    pack_xt_kernel<<<dim3((NN * HIDD + 255) / 256), dim3(256), 0, stream>>>(
        Xs, Xt + (size_t)t * NN * 32, Mt + (size_t)t * NN * 32, xcat);
    gemm(xcat, Wi_pk, inB, nullptr, xt, NN, HIDD, HIDD, GF_BIAS);

    run_conv(Wq_d, (const float*)d_in[CD + 6], Wk_d, (const float*)d_in[CD + 5],
             Wv_d, (const float*)d_in[CD + 8], Ws_d, (const float*)d_in[CD + 7],
             (const float*)d_in[CD + 0], ei_d, ei_d + EE, ea_d, hd);
    run_conv(Wq_u, (const float*)d_in[CU + 6], Wk_u, (const float*)d_in[CU + 5],
             Wv_u, (const float*)d_in[CU + 8], Ws_u, (const float*)d_in[CU + 7],
             (const float*)d_in[CU + 0], ei_u, ei_u + EE, ea_u, hu);

    // h = relu(concat(hd,hu) @ Wl + bl), split into two K=512 GEMMs
    float* ht = Hs + (size_t)t * NN * HIDD;
    gemm(hd, Wl_lo, linB, nullptr, ht, NN, HC, HIDD, GF_BIAS);
    gemm(hu, Wl_hi, nullptr, ht, ht, NN, HC, HIDD, GF_RES | GF_RELU);
  }

  // -------- time encoding + transpose to [N,T,128] --------
  time_enc_kernel<<<dim3(1), dim3(256), 0, stream>>>(yrs, te);
  gemm(te, Wt_pk, tB, nullptr, tpj, TT, 32, HIDD, GF_BIAS);
  add_time_transpose_kernel<<<dim3((unsigned)(((size_t)TT * NN * HIDD + 255) / 256)),
                              dim3(256), 0, stream>>>(Hs, tpj, xb);

  // -------- transformer encoder (2 layers) --------
  for (int L = 0; L < 2; ++L) {
    int bidx = 28 + 12 * L;
    const float* b1    = (const float*)d_in[bidx + 4];
    const float* b2    = (const float*)d_in[bidx + 5];
    const float* beta1 = (const float*)d_in[bidx + 6];
    const float* beta2 = (const float*)d_in[bidx + 7];
    const float* bo    = (const float*)d_in[bidx + 8];
    const float* bqkv  = (const float*)d_in[bidx + 9];
    const float* g1    = (const float*)d_in[bidx + 10];
    const float* g2    = (const float*)d_in[bidx + 11];

    gemm(xb, Wqkv_pk[L], bqkv, nullptr, qkvb, NTOK, HIDD, 384, GF_BIAS);
    mha_attn_kernel<<<dim3(NN), dim3(128), 0, stream>>>(qkvb, attno);
    gemm(attno, Wo_pk[L], bo, xb, tmpb, NTOK, HIDD, HIDD, GF_BIAS | GF_RES);
    layer_norm_kernel<<<dim3((NTOK + 7) / 8), dim3(256), 0, stream>>>(tmpb, g1, beta1, xb, NTOK);
    gemm(xb, W1_pk[L], b1, nullptr, ffh, NTOK, HIDD, 256, GF_BIAS | GF_RELU);
    gemm(ffh, W2_pk[L], b2, xb, tmpb, NTOK, 256, HIDD, GF_BIAS | GF_RES);
    layer_norm_kernel<<<dim3((NTOK + 7) / 8), dim3(256), 0, stream>>>(tmpb, g2, beta2, xb, NTOK);
  }

  // -------- decay-weighted pooling + readout --------
  compute_w_kernel<<<dim3(1), dim3(32), 0, stream>>>(
      yrs, wbuf, (float*)d_out + (size_t)NN * HIDD);
  weighted_sum_kernel<<<dim3((NN * HIDD + 255) / 256), dim3(256), 0, stream>>>(xb, wbuf, zb);
  gemm(zb, Wr_pk, roB, nullptr, (float*)d_out, NN, HIDD, HIDD, GF_BIAS);
}
